// sLSTM_76836964925455
// MI455X (gfx1250) — matrix operand order repaired and checked by reading, not compiled, not run on hardware
//
#include <hip/hip_runtime.h>

// sLSTM for MI455X (gfx1250, wave32, WMMA + TDM).
// B=64, T=512, N=1024. bf16 WMMA (v_wmma_f32_16x16x32_bf16) with f32 accum.
// Phase 1: xz = x @ W_x + b_x   (32768x1024 @ 1024x4096), stored f32 in ws.
// Phase 2: 512 fused step kernels: z = xz_t + h@W_h + b_h; gates; update c,h.
//          One 16x16 (b,n) tile per block; wave g computes gate g.
//          h-tile staged to LDS once per block via TENSOR_LOAD_TO_LDS (TDM),
//          with TDM padding (4 DWORDs per 256) to skew LDS banks.
// Phase 3: hidden = h @ W_out + b_out -> d_out[0:BN); c copied to d_out[BN:2BN).

#define B_  64
#define T_  512
#define N_  1024
#define GM_ 4096          // 4*N
#define R_  (T_ * B_)     // 32768 rows of the xz GEMM
#define HROW_ 1040        // padded LDS row stride in bf16 (1024 data + 2x8 pad)

typedef __bf16 bf16;
typedef __attribute__((ext_vector_type(16))) __bf16 bf16x16;
typedef __attribute__((ext_vector_type(8)))  __bf16 bf16x8;
typedef __attribute__((ext_vector_type(8)))  float  f32x8;
typedef __attribute__((ext_vector_type(4)))  unsigned int u32x4;
typedef __attribute__((ext_vector_type(8)))  int i32x8;
typedef __attribute__((ext_vector_type(4)))  int i32x4;

#if defined(__has_builtin)
#if __has_builtin(__builtin_amdgcn_tensor_load_to_lds)
#define HAVE_TDM 1
#endif
#endif
#ifndef HAVE_TDM
#define HAVE_TDM 0
#endif

__device__ __forceinline__ bf16x16 ld_frag_pair(const bf16* p0, const bf16* p1) {
    bf16x8 lo = *(const bf16x8*)p0;
    bf16x8 hi = *(const bf16x8*)p1;
    bf16x16 r;
#pragma unroll
    for (int i = 0; i < 8; ++i) { r[i] = lo[i]; r[i + 8] = hi[i]; }
    return r;
}

// A fragment (16x32 bf16, MxK), row-major source with leading dim lda.
// lane&15 = row; lanes 0-15 hold k {0..7, 16..23}, lanes 16-31 hold {8..15, 24..31}.
__device__ __forceinline__ bf16x16 ld_fragA(const bf16* __restrict__ A, int lda,
                                            int row, int k, int lane) {
    const bf16* p = A + (long)row * lda + k + ((lane >> 4) * 8);
    return ld_frag_pair(p, p + 16);
}

// A fragment from the padded LDS tile (row stride HROW_, +8 elems after k>=512).
// k is a multiple of 32 so a fragment never straddles a pad boundary.
__device__ __forceinline__ bf16x16 ld_fragA_lds(const bf16* sh, int row, int k, int lane) {
    const bf16* p = sh + row * HROW_ + k + 8 * (k >> 9) + ((lane >> 4) * 8);
    return ld_frag_pair(p, p + 16);
}

// B fragment (32x16 bf16, KxN) from column-major storage Bt[col][k] (ldb along k).
// lane&15 = col; lanes 0-15 hold k {0..7, 16..23}, lanes 16-31 hold {8..15, 24..31}.
__device__ __forceinline__ bf16x16 ld_fragB(const bf16* __restrict__ Bt, int ldb,
                                            int col, int k, int lane) {
    const bf16* p = Bt + (long)col * ldb + k + ((lane >> 4) * 8);
    return ld_frag_pair(p, p + 16);
}

// ---- conversion kernels -----------------------------------------------------

// xb[(t*B + b)*N + n] = bf16( x[(b*T + t)*N + n] )
__global__ void cvt_x_kernel(const float* __restrict__ x, bf16* __restrict__ xb) {
    long total = (long)R_ * N_;
    for (long idx = (long)blockIdx.x * blockDim.x + threadIdx.x; idx < total;
         idx += (long)gridDim.x * blockDim.x) {
        int  n = (int)(idx & (N_ - 1));
        long r = idx >> 10;            // r = t*B + b
        int  b = (int)(r & (B_ - 1));
        int  t = (int)(r >> 6);
        xb[idx] = (bf16)x[((long)b * T_ + t) * N_ + n];
    }
}

// W[g][n][m] (f32) -> Wt[g][m][n] (bf16), G groups of N x N
__global__ void cvt_wT_kernel(const float* __restrict__ W, bf16* __restrict__ Wt, int G) {
    long total = (long)G * N_ * N_;
    for (long idx = (long)blockIdx.x * blockDim.x + threadIdx.x; idx < total;
         idx += (long)gridDim.x * blockDim.x) {
        int n = (int)(idx & (N_ - 1));
        int m = (int)((idx >> 10) & (N_ - 1));
        int g = (int)(idx >> 20);
        Wt[idx] = (bf16)W[(((long)g << 10) + n) * N_ + m];
    }
}

__global__ void init_kernel(const float* __restrict__ h_in,
                            bf16* __restrict__ hb0, float* __restrict__ c) {
    int idx = blockIdx.x * blockDim.x + threadIdx.x;
    if (idx < B_ * N_) {
        c[idx]   = 0.0f;
        hb0[idx] = (bf16)h_in[idx];
    }
}

// ---- phase 1: xz = xb @ Wx^T + b_x ------------------------------------------
// grid (GM/64, R/64), block 128 (4 waves). Wave computes 16 rows x 64 cols.
__global__ void __launch_bounds__(128)
gemm_xz_kernel(const bf16* __restrict__ xb, const bf16* __restrict__ Wxt,
               const float* __restrict__ bx, float* __restrict__ xz) {
    int lane = threadIdx.x & 31;
    int wave = threadIdx.x >> 5;
    int row0 = blockIdx.y * 64 + wave * 16;
    int col0 = blockIdx.x * 64;

    f32x8 acc[4] = {};
    int arow = row0 + (lane & 15);
    int ccol = lane & 15;

    for (int k = 0; k < N_; k += 32) {
        bf16x16 a = ld_fragA(xb, N_, arow, k, lane);
#pragma unroll
        for (int j = 0; j < 4; ++j) {
            bf16x16 b = ld_fragB(Wxt, N_, col0 + j * 16 + ccol, k, lane);
            acc[j] = __builtin_amdgcn_wmma_f32_16x16x32_bf16(
                false, a, false, b, (short)0, acc[j], false, false);
        }
    }

    int rl = 8 * (lane >> 4);
#pragma unroll
    for (int j = 0; j < 4; ++j) {
        int   col  = col0 + j * 16 + ccol;
        float bias = bx[col];
#pragma unroll
        for (int v = 0; v < 8; ++v) {
            int row = row0 + rl + v;
            xz[(long)row * GM_ + col] = acc[j][v] + bias;
        }
    }
}

// ---- phase 2: fused recurrence step -----------------------------------------
// grid (N/16, B/16), block 128. Wave g computes gate g's 16x16 z-tile.
__global__ void __launch_bounds__(128)
slstm_step_kernel(const bf16* __restrict__ h_in, bf16* __restrict__ h_out,
                  const bf16* __restrict__ Wht, const float* __restrict__ bh,
                  const float* __restrict__ xz_t, float* __restrict__ c) {
    __shared__ bf16  hsh[16 * HROW_];   // padded 16x1024 h tile (32.5 KB)
    __shared__ float zs[4][16][17];

    int lane = threadIdx.x & 31;
    int g    = threadIdx.x >> 5;
    int b0   = blockIdx.y * 16;
    int n0   = blockIdx.x * 16;

#if HAVE_TDM
    if (g == 0) {
        // Tensor DMA: 2-D tile (16 rows x 1024 cols, 2B elems) of the 64x1024
        // h matrix -> LDS, with pad_amount=4 DWORDs every 256 DWORDs
        // (row stride in LDS becomes 1040 elems => bank-skewed A rows).
        unsigned long long ga =
            (unsigned long long)(const void*)(h_in + (unsigned long long)b0 * N_);
        unsigned int lds = (unsigned int)(unsigned long long)(const void*)hsh;
        u32x4 d0;
        d0[0] = 1u;                                             // count=1 (valid D#)
        d0[1] = lds;                                            // lds_addr
        d0[2] = (unsigned int)ga;                               // global_addr[31:0]
        d0[3] = (unsigned int)((ga >> 32) & 0x1FFFFFFu)         // global_addr[56:32]
              | (2u << 30);                                     // type=2 (image)
        i32x8 d1;
        d1[0] = (int)((1u << 16)     // data_size = 1 (2 bytes)
                    | (1u << 20)     // pad_enable
                    | (7u << 22)     // pad_interval: every 256 DWORDs
                    | (3u << 25));   // pad_amount: 4 DWORDs
        d1[1] = (int)(1024u << 16);  // tensor_dim0 = 1024 (bits 79:48)
        d1[2] = (int)(64u << 16);    // tensor_dim1 = 64   (bits 111:80)
        d1[3] = (int)(1024u << 16);  // tile_dim0   = 1024 (bits 127:112)
        d1[4] = 16;                  // tile_dim1   = 16   (bits 143:128)
        d1[5] = 1024;                // tensor_dim0_stride = 1024 (bits 207:160)
        d1[6] = 0;
        d1[7] = 0;
#if __clang_major__ >= 23
        __builtin_amdgcn_tensor_load_to_lds(d0, d1, (i32x4)0, (i32x4)0, (i32x8)0, 0);
#else
        __builtin_amdgcn_tensor_load_to_lds(d0, d1, (i32x4)0, (i32x4)0, 0);
#endif
        __builtin_amdgcn_s_wait_tensorcnt(0);
    }
#else
    // Fallback: cooperative copy into the identical padded layout.
    for (int ci = threadIdx.x; ci < 2048; ci += 128) {
        int row = ci >> 7;
        int k   = (ci & 127) * 8;
        bf16x8 v = *(const bf16x8*)(h_in + (long)(b0 + row) * N_ + k);
        *(bf16x8*)(hsh + row * HROW_ + k + 8 * (k >> 9)) = v;
    }
#endif
    __syncthreads();

    f32x8 acc = {};
    int bcol = n0 + (lane & 15);
    const bf16* Wg = Wht + (long)g * N_ * N_;

    for (int k = 0; k < N_; k += 32) {
        bf16x16 a = ld_fragA_lds(hsh, lane & 15, k, lane);
        bf16x16 b = ld_fragB(Wg, N_, bcol, k, lane);
        acc = __builtin_amdgcn_wmma_f32_16x16x32_bf16(
            false, a, false, b, (short)0, acc, false, false);
    }

    int   rl   = 8 * (lane >> 4);
    int   nl   = lane & 15;
    float bias = bh[g * N_ + n0 + nl];
#pragma unroll
    for (int v = 0; v < 8; ++v) {
        int bg = b0 + rl + v;
        zs[g][rl + v][nl] = acc[v] + xz_t[(long)bg * GM_ + g * N_ + n0 + nl] + bias;
    }
    __syncthreads();

    // 256 tile elements over 128 threads
    for (int e = threadIdx.x; e < 256; e += 128) {
        int bl = e >> 4, nn = e & 15;
        float zi = zs[0][bl][nn], zf = zs[1][bl][nn];
        float zc = zs[2][bl][nn], zo = zs[3][bl][nn];
        float iv = __expf(zi);
        float fv = __expf(zf);
        long  idx = (long)(b0 + bl) * N_ + n0 + nn;
        float cv  = c[idx];
        float cn  = (fv * cv + iv * tanhf(zc)) / (fv + iv);
        float o   = 1.0f / (1.0f + __expf(-zo));
        c[idx]    = cn;
        h_out[idx] = (bf16)(o * tanhf(cn));
    }
}

// ---- phase 3: hidden = h @ W_out^T + b_out ----------------------------------
// grid (N/64), block 128 (4 waves cover the 64 rows).
__global__ void __launch_bounds__(128)
gemm_out_kernel(const bf16* __restrict__ hb, const bf16* __restrict__ Wot,
                const float* __restrict__ bo, float* __restrict__ out) {
    int lane = threadIdx.x & 31;
    int wave = threadIdx.x >> 5;
    int row0 = wave * 16;
    int col0 = blockIdx.x * 64;

    f32x8 acc[4] = {};
    int arow = row0 + (lane & 15);
    int ccol = lane & 15;

    for (int k = 0; k < N_; k += 32) {
        bf16x16 a = ld_fragA(hb, N_, arow, k, lane);
#pragma unroll
        for (int j = 0; j < 4; ++j) {
            bf16x16 b = ld_fragB(Wot, N_, col0 + j * 16 + ccol, k, lane);
            acc[j] = __builtin_amdgcn_wmma_f32_16x16x32_bf16(
                false, a, false, b, (short)0, acc[j], false, false);
        }
    }

    int rl = 8 * (lane >> 4);
#pragma unroll
    for (int j = 0; j < 4; ++j) {
        int   col  = col0 + j * 16 + ccol;
        float bias = bo[col];
#pragma unroll
        for (int v = 0; v < 8; ++v) {
            int row = row0 + rl + v;
            out[(long)row * N_ + col] = acc[j][v] + bias;
        }
    }
}

// ---- host -------------------------------------------------------------------

extern "C" void kernel_launch(void* const* d_in, const int* in_sizes, int n_in,
                              void* d_out, int out_size, void* d_ws, size_t ws_size,
                              hipStream_t stream) {
    const float* x   = (const float*)d_in[0];
    const float* h0  = (const float*)d_in[1];
    const float* W_x = (const float*)d_in[2];
    const float* b_x = (const float*)d_in[3];
    const float* W_h = (const float*)d_in[4];
    const float* b_h = (const float*)d_in[5];
    const float* W_o = (const float*)d_in[6];
    const float* b_o = (const float*)d_in[7];
    float* out = (float*)d_out;
    (void)in_sizes; (void)n_in; (void)out_size; (void)ws_size;

    char*  ws  = (char*)d_ws;
    size_t off = 0;
    auto alloc = [&](size_t bytes) -> char* {
        char* p = ws + off;
        off = (off + bytes + 255) & ~(size_t)255;
        return p;
    };

    bf16*  xb   = (bf16*)alloc((size_t)R_ * N_ * 2);       //  64 MB
    bf16*  Wxt  = (bf16*)alloc((size_t)4 * N_ * N_ * 2);   //   8 MB
    bf16*  Wht  = (bf16*)alloc((size_t)4 * N_ * N_ * 2);   //   8 MB
    bf16*  Wot  = (bf16*)alloc((size_t)N_ * N_ * 2);       //   2 MB
    bf16*  hbuf = (bf16*)alloc((size_t)2 * B_ * N_ * 2);   // 256 KB (double buffer)
    float* cws  = (float*)alloc((size_t)B_ * N_ * 4);      // 256 KB
    float* xz   = (float*)alloc((size_t)R_ * GM_ * 4);     // 512 MB

    cvt_x_kernel <<<2048, 256, 0, stream>>>(x, xb);
    cvt_wT_kernel<<<2048, 256, 0, stream>>>(W_x, Wxt, 4);
    cvt_wT_kernel<<<2048, 256, 0, stream>>>(W_h, Wht, 4);
    cvt_wT_kernel<<<512,  256, 0, stream>>>(W_o, Wot, 1);
    init_kernel  <<<(B_ * N_) / 256, 256, 0, stream>>>(h0, hbuf, cws);

    gemm_xz_kernel<<<dim3(GM_ / 64, R_ / 64), 128, 0, stream>>>(xb, Wxt, b_x, xz);

    for (int t = 0; t < T_; ++t) {
        const bf16* hin  = hbuf + (size_t)(t & 1) * B_ * N_;
        bf16*       hout = hbuf + (size_t)((t + 1) & 1) * B_ * N_;
        slstm_step_kernel<<<dim3(N_ / 16, B_ / 16), 128, 0, stream>>>(
            hin, hout, Wht, b_h, xz + (size_t)t * B_ * GM_, cws);
    }

    // After T=512 steps the live h buffer is index (T & 1) == 0.
    gemm_out_kernel<<<dim3(N_ / 64), 128, 0, stream>>>(hbuf, Wot, b_o, out);

    hipMemcpyAsync(out + (size_t)B_ * N_, cws, (size_t)B_ * N_ * 4,
                   hipMemcpyDeviceToDevice, stream);
}
